// BasicBlock_73272142069886
// MI455X (gfx1250) — matrix operand order repaired and checked by reading, not compiled, not run on hardware
//
#include <hip/hip_runtime.h>

typedef __attribute__((ext_vector_type(16))) _Float16 v16h;
typedef __attribute__((ext_vector_type(8)))  _Float16 v8h;
typedef __attribute__((ext_vector_type(8)))  float    v8f;

#define W_IMG   112
#define HW_IMG  12544                    // 112*112
#define C_IN    64
#define C_OUT   64
#define N_IMG   32
#define M_TOT   (N_IMG * HW_IMG)         // 401408 spatial positions
#define H_PAD   114
#define W_PAD   114
#define PIX_PAD (H_PAD * W_PAD)          // 12996 pixels per padded image
#define PAD_HALVES ((size_t)N_IMG * PIX_PAD * C_IN)   // 26,615,808 halves
#define FRAG_HALVES 512                  // one B fragment: 32 lanes * 16 halves
#define LAYER_WQ_HALVES 36864            // 9 taps * 2 ci-blocks * 4 n-tiles * 512
#define M_TILES 4                        // vertically stacked 16-pixel rows per wave
#define WT_PER_ROW 7                     // 112 / 16 w-tiles per row
#define HG_PER_IMG 28                    // 112 / 4 h-groups per image
#define DELTA_T 0.3f
#define EPS_BN  1e-5f

union V16U { v16h v; v8h h[2]; };

// ---------------------------------------------------------------------------
// Ternarize weights and emit them in the contiguous-k B-fragment layout used here for
// v_wmma_f32_16x16x32_f16 (lane L: column N = L%16; lanes 0-15 hold K=0..15,
// lanes 16-31 hold K=16..31, 16 consecutive halves per lane). Also zeroes the
// BN statistics accumulators (256 floats) each call.
// ---------------------------------------------------------------------------
__global__ __launch_bounds__(256)
void prep_kernel(const float* __restrict__ w1, const float* __restrict__ w2,
                 _Float16* __restrict__ wq, float* __restrict__ stats)
{
    int idx = blockIdx.x * 256 + threadIdx.x;
    if (idx < 256) stats[idx] = 0.0f;               // sum1, sq1, sum2, sq2
    if (idx >= 2 * 9 * 2 * 4 * 32) return;          // 4608 producer threads

    int layer = idx / 2304;
    int rem   = idx % 2304;
    int p     = rem / 256;                           // kernel tap 0..8
    int rem2  = rem % 256;
    int b     = rem2 / 128;                          // ci block 0..1
    int rem3  = rem2 % 128;
    int tile  = rem3 / 32;                           // n-tile 0..3
    int lane  = rem3 % 32;

    const float* w = layer ? w2 : w1;
    int co    = tile * 16 + (lane & 15);
    int kbase = b * 32 + ((lane >= 16) ? 16 : 0);    // absolute ci of half 0
    int kh    = p / 3, kw = p % 3;

    _Float16* dst = wq + (size_t)layer * LAYER_WQ_HALVES
                  + ((size_t)((p * 2 + b) * 4 + tile) * 32 + lane) * 16;
    #pragma unroll
    for (int j = 0; j < 16; ++j) {
        int ci  = kbase + j;
        float v = w[((co * C_IN + ci) * 3 + kh) * 3 + kw];
        float q = (fabsf(v) > DELTA_T) ? (v > 0.0f ? 1.0f : -1.0f) : 0.0f;
        dst[j]  = (_Float16)q;
    }
}

// ---------------------------------------------------------------------------
// Zero the entire padded activation buffer (32B per thread). Interior is then
// overwritten by to_nhwc / bn_act; the halo stays zero for both convs.
// ---------------------------------------------------------------------------
__global__ __launch_bounds__(256)
void zero_pad_kernel(_Float16* __restrict__ buf)
{
    size_t i = ((size_t)blockIdx.x * 256 + threadIdx.x) * 16;  // halves
    if (i >= PAD_HALVES) return;
    const v16h z = {};
    *(v16h*)(buf + i) = z;
}

// ---------------------------------------------------------------------------
// x (NCHW f32) -> padded activation (NHWC f16, 114x114 with zero halo)
// ---------------------------------------------------------------------------
__global__ __launch_bounds__(256)
void to_nhwc_f16_kernel(const float* __restrict__ x, _Float16* __restrict__ act)
{
    size_t idx = (size_t)blockIdx.x * 256 + threadIdx.x;
    if (idx >= (size_t)N_IMG * C_OUT * HW_IMG) return;
    int pos = (int)(idx % HW_IMG);
    int c   = (int)((idx / HW_IMG) % C_OUT);
    int n   = (int)(idx / ((size_t)HW_IMG * C_OUT));
    int h   = pos / W_IMG;
    int w   = pos - h * W_IMG;
    size_t dst = ((size_t)n * PIX_PAD + (size_t)(h + 1) * W_PAD + (w + 1)) * C_IN + c;
    act[dst] = (_Float16)x[idx];
}

// ---------------------------------------------------------------------------
// Implicit-GEMM 3x3 conv via WMMA on padded NHWC activations. One wave
// computes a 64(M)x64(N) macro-tile laid out as 4 vertically-stacked
// 16-pixel rows x 64 output channels. K loop: 9 taps x 2 ci-blocks; per step
// 4 B-frags load once and are reused by 4 A-frags -> 16 WMMA (288 total).
// All A addresses are compile-time immediate offsets from ONE per-lane base
// (tile mt shifts by mt*W_PAD*C_IN halves), so the inner loop is pure
// clause-loads + WMMA with no address arithmetic; vertically-overlapping taps
// are CSE-able identical loads. Halo padding supplies boundary zeros.
// A fragment (contiguous-k order, same as B): lane L = row M=L%16; lanes 0-15 hold
// K 0..15, lanes 16-31 hold K 16..31 -> two 16B NHWC loads.
// ---------------------------------------------------------------------------
__global__ __launch_bounds__(256)
void conv3x3_wmma_kernel(const _Float16* __restrict__ act,     // padded NHWC f16
                         const _Float16* __restrict__ wq,      // frag-ready, this layer
                         const float*    __restrict__ residual,// NCHW f32 or null
                         _Float16*       __restrict__ outPre,  // NHWC f16 (pre-BN)
                         float* __restrict__ gsum,
                         float* __restrict__ gsq)
{
    __shared__ float s_sum[C_OUT];
    __shared__ float s_sq[C_OUT];
    int tid = threadIdx.x;
    if (tid < C_OUT)            s_sum[tid] = 0.0f;
    else if (tid < 2 * C_OUT)   s_sq[tid - C_OUT] = 0.0f;
    __syncthreads();

    const int wave = tid >> 5;
    const int lane = tid & 31;
    const int r    = lane & 15;
    const int hi   = lane >> 4;
    const int kA0  = hi ? 16 : 0;

    // Macro-tile decomposition: one division chain per wave.
    const int wid = blockIdx.x * 8 + wave;
    const int wt  = wid % WT_PER_ROW;            // w-tile in row
    const int t1  = wid / WT_PER_ROW;
    const int hg  = t1 % HG_PER_IMG;             // h-group (4 rows)
    const int n   = t1 / HG_PER_IMG;             // image
    const int h   = hg * M_TILES;                // first row of group
    const int w0  = wt * 16;                     // first pixel of row-tile
    const int m0  = n * HW_IMG + h * W_IMG + w0; // first output position

    // Single per-lane A base; every K-loop access is an immediate offset.
    const _Float16* ap = act
        + ((size_t)n * PIX_PAD + (size_t)(h + 1) * W_PAD + (w0 + r + 1)) * C_IN + kA0;
    const _Float16* bp = wq + (size_t)lane * 16;

    v8f acc[M_TILES][4];
    #pragma unroll
    for (int mt = 0; mt < M_TILES; ++mt)
        #pragma unroll
        for (int t = 0; t < 4; ++t)
            acc[mt][t] = (v8f){};

    #pragma unroll
    for (int p = 0; p < 9; ++p) {
        const int doff = ((p / 3 - 1) * W_PAD + (p % 3 - 1)) * C_IN;
        #pragma unroll
        for (int b = 0; b < 2; ++b) {
            const _Float16* bpp = bp + (size_t)((p * 2 + b) * 4) * FRAG_HALVES;
            const v16h b0 = *(const v16h*)(bpp);
            const v16h b1 = *(const v16h*)(bpp + 1 * FRAG_HALVES);
            const v16h b2 = *(const v16h*)(bpp + 2 * FRAG_HALVES);
            const v16h b3 = *(const v16h*)(bpp + 3 * FRAG_HALVES);
            #pragma unroll
            for (int mt = 0; mt < M_TILES; ++mt) {
                const int aoff = doff + mt * (W_PAD * C_IN) + b * 32;  // compile-time
                V16U a;
                a.h[0] = *(const v8h*)(ap + aoff);
                a.h[1] = *(const v8h*)(ap + aoff + 8);
                acc[mt][0] = __builtin_amdgcn_wmma_f32_16x16x32_f16(false, a.v, false, b0, (short)0, acc[mt][0], false, false);
                acc[mt][1] = __builtin_amdgcn_wmma_f32_16x16x32_f16(false, a.v, false, b1, (short)0, acc[mt][1], false, false);
                acc[mt][2] = __builtin_amdgcn_wmma_f32_16x16x32_f16(false, a.v, false, b2, (short)0, acc[mt][2], false, false);
                acc[mt][3] = __builtin_amdgcn_wmma_f32_16x16x32_f16(false, a.v, false, b3, (short)0, acc[mt][3], false, false);
            }
        }
    }

    // Epilogue: residual add (optional), pre-BN f16 store, LDS stat reduce.
    // D layout: VGPR v holds row M = v + 8*hi, column N (=co) = lane%16.
    // Per-mt bases are constant offsets from the macro-tile base.
    _Float16* opBase = outPre + (size_t)(m0 + 8 * hi) * C_OUT;
    const float* rpBase = residual
        ? residual + (size_t)n * (C_OUT * HW_IMG) + (size_t)h * W_IMG + (w0 + 8 * hi)
        : nullptr;

    #pragma unroll
    for (int mt = 0; mt < M_TILES; ++mt) {
        #pragma unroll
        for (int t = 0; t < 4; ++t) {
            const int co = t * 16 + r;
            float ls = 0.0f, lq = 0.0f;
            _Float16* op = opBase + (size_t)mt * (W_IMG * C_OUT) + co;
            const float* rp = residual
                ? rpBase + (size_t)co * HW_IMG + (size_t)mt * W_IMG
                : nullptr;
            #pragma unroll
            for (int v = 0; v < 8; ++v) {
                float val = acc[mt][t][v];
                if (residual) val += rp[v];
                op[(size_t)v * C_OUT] = (_Float16)val;
                ls += val;
                lq += val * val;
            }
            atomicAdd(&s_sum[co], ls);
            atomicAdd(&s_sq[co], lq);
        }
    }

    __syncthreads();
    if (tid < C_OUT)          atomicAdd(&gsum[tid], s_sum[tid]);
    else if (tid < 2 * C_OUT) atomicAdd(&gsq[tid - C_OUT], s_sq[tid - C_OUT]);
}

// ---------------------------------------------------------------------------
// Per-channel BN scale/shift from accumulated batch statistics.
// ---------------------------------------------------------------------------
__global__ void bn_finalize_kernel(const float* __restrict__ gsum,
                                   const float* __restrict__ gsq,
                                   const float* __restrict__ gamma,
                                   const float* __restrict__ beta,
                                   float* __restrict__ scale,
                                   float* __restrict__ shift)
{
    int c = threadIdx.x;
    if (c >= C_OUT) return;
    const float invM = 1.0f / (float)M_TOT;
    float mean = gsum[c] * invM;
    float var  = gsq[c] * invM - mean * mean;
    float sc   = rsqrtf(var + EPS_BN) * gamma[c];
    scale[c] = sc;
    shift[c] = beta[c] - mean * sc;
}

// ---------------------------------------------------------------------------
// BN + hardtanh: pre-BN f16 NHWC (unpadded) -> padded NHWC f16 (conv2 input).
// Interior only; halo remains zero from zero_pad_kernel earlier in this call.
// ---------------------------------------------------------------------------
__global__ __launch_bounds__(256)
void bn_act_kernel(const _Float16* __restrict__ pre,
                   const float* __restrict__ scale,
                   const float* __restrict__ shift,
                   _Float16* __restrict__ act)
{
    size_t idx = (size_t)blockIdx.x * 256 + threadIdx.x;
    if (idx >= (size_t)M_TOT * C_OUT) return;
    int c = (int)(idx & (C_OUT - 1));
    int m = (int)(idx >> 6);
    int n = m / HW_IMG;
    int pos = m - n * HW_IMG;
    int h = pos / W_IMG;
    int w = pos - h * W_IMG;
    float v = (float)pre[idx];
    float y = fminf(fmaxf(v * scale[c] + shift[c], -1.0f), 1.0f);
    size_t dst = ((size_t)n * PIX_PAD + (size_t)(h + 1) * W_PAD + (w + 1)) * C_IN + c;
    act[dst] = (_Float16)y;
}

// ---------------------------------------------------------------------------
// Final BN + hardtanh, NHWC f16 (unpadded) -> NCHW f32 output.
// ---------------------------------------------------------------------------
__global__ __launch_bounds__(256)
void final_kernel(const _Float16* __restrict__ pre,
                  const float* __restrict__ scale,
                  const float* __restrict__ shift,
                  float* __restrict__ out)
{
    size_t idx = (size_t)blockIdx.x * 256 + threadIdx.x;
    if (idx >= (size_t)N_IMG * C_OUT * HW_IMG) return;
    int pos = (int)(idx % HW_IMG);
    int c   = (int)((idx / HW_IMG) % C_OUT);
    int n   = (int)(idx / ((size_t)HW_IMG * C_OUT));
    float v = (float)pre[((size_t)n * HW_IMG + pos) * C_OUT + c];
    out[idx] = fminf(fmaxf(v * scale[c] + shift[c], -1.0f), 1.0f);
}

// ---------------------------------------------------------------------------
extern "C" void kernel_launch(void* const* d_in, const int* in_sizes, int n_in,
                              void* d_out, int out_size, void* d_ws, size_t ws_size,
                              hipStream_t stream)
{
    (void)in_sizes; (void)n_in; (void)out_size; (void)ws_size;
    const float* x  = (const float*)d_in[0];
    const float* w1 = (const float*)d_in[1];
    const float* g1 = (const float*)d_in[2];
    const float* b1 = (const float*)d_in[3];
    const float* w2 = (const float*)d_in[4];
    const float* g2 = (const float*)d_in[5];
    const float* b2 = (const float*)d_in[6];
    float* out = (float*)d_out;

    char* ws = (char*)d_ws;
    const size_t PAD_BYTES = PAD_HALVES * sizeof(_Float16);            // 53,231,616
    const size_t ACT_BYTES = (size_t)M_TOT * C_IN * sizeof(_Float16);  // 51,380,224
    _Float16* bufA = (_Float16*)ws;                                    // padded activations
    _Float16* bufB = (_Float16*)(ws + PAD_BYTES);                      // pre-BN (unpadded)
    _Float16* wq   = (_Float16*)(ws + PAD_BYTES + ACT_BYTES);          // frag-ready weights
    float* stats   = (float*)(ws + PAD_BYTES + ACT_BYTES
                              + 2 * LAYER_WQ_HALVES * sizeof(_Float16)); // 256 f32
    float* scales  = stats + 256;                                        // 256 f32

    const int TOT     = N_IMG * C_OUT * HW_IMG;        // 25,690,112 (multiple of 256)
    const int EBLK    = TOT / 256;                     // 100352
    const int CONVBLK = M_TOT / (16 * M_TILES * 8);    // 784 (8 waves * 64 px each)
    const int ZBLK    = (int)(PAD_HALVES / 16 / 256);  // 6498 (exact)

    // 1) ternarize weights into fragment layout, zero BN stats
    prep_kernel<<<18, 256, 0, stream>>>(w1, w2, wq, stats);
    // 2) zero padded activation buffer (halo stays zero for both convs)
    zero_pad_kernel<<<ZBLK, 256, 0, stream>>>(bufA);
    // 3) x -> padded NHWC f16
    to_nhwc_f16_kernel<<<EBLK, 256, 0, stream>>>(x, bufA);
    // 4) conv1 (WMMA) -> pre1 (bufB) + stats1
    conv3x3_wmma_kernel<<<CONVBLK, 256, 0, stream>>>(bufA, wq, nullptr, bufB,
                                                     stats, stats + 64);
    // 5) BN1 scale/shift
    bn_finalize_kernel<<<1, 64, 0, stream>>>(stats, stats + 64, g1, b1,
                                             scales, scales + 64);
    // 6) BN1 + hardtanh -> padded act2 (bufA interior rewritten)
    bn_act_kernel<<<EBLK, 256, 0, stream>>>(bufB, scales, scales + 64, bufA);
    // 7) conv2 (WMMA) + residual -> pre2 (bufB reused) + stats2
    conv3x3_wmma_kernel<<<CONVBLK, 256, 0, stream>>>(bufA, wq + LAYER_WQ_HALVES, x,
                                                     bufB, stats + 128, stats + 192);
    // 8) BN2 scale/shift
    bn_finalize_kernel<<<1, 64, 0, stream>>>(stats + 128, stats + 192, g2, b2,
                                             scales + 128, scales + 192);
    // 9) BN2 + hardtanh -> f32 NCHW output
    final_kernel<<<EBLK, 256, 0, stream>>>(bufB, scales + 128, scales + 192, out);
}
